// RNNModel_47278999995119
// MI455X (gfx1250) — matrix operand order repaired and checked by reading, not compiled, not run on hardware
//
#include <hip/hip_runtime.h>

#define TEMPC 65.0f
#define EPSL  1e-6f

enum : int {
  kNTOK = 33278, kNINP = 400, kNHID = 1150,
  kS = 70, kB = 128, kNS = 10,
  kKP = 416,            // NINP padded to 13*32
  kHP = 1152,           // NHID padded to 72*16 (= 36*32)
  kNPOS = kS * kB,      // 8960
  kMT = kNPOS / 16,     // 560 M-tiles of positions
  kCT = kHP / 16,       // 72 N-tiles of hidden units
  kKC = kKP / 32        // 13 K-chunks for the input projection
};

typedef __attribute__((ext_vector_type(16))) __bf16 v16bf;
typedef __attribute__((ext_vector_type(8)))  float  v8f;
typedef __attribute__((ext_vector_type(4)))  unsigned int u32x4;

__device__ __forceinline__ unsigned short f2bf(float f) {
  unsigned int u = __float_as_uint(f);
  u += 0x7FFFu + ((u >> 16) & 1u);        // round-to-nearest-even
  return (unsigned short)(u >> 16);
}

// A fragment: 16x32 bf16 from row-major source; rowp = this lane's row
__device__ __forceinline__ v16bf load_a(const unsigned short* rowp, int k0, int lane) {
  const unsigned short* p = rowp + k0 + ((lane & 16) ? 8 : 0);
  v16bf a;
  ((u32x4*)&a)[0] = *(const u32x4*)(p);        // K: +0..+7
  ((u32x4*)&a)[1] = *(const u32x4*)(p + 16);   // K: +16..+23
  return a;
}

// B fragment: 32x16 bf16 from weights stored [N][K] row-major (B[k][n] = W[n][k])
__device__ __forceinline__ v16bf load_b(const unsigned short* W, int ldk, int n0, int k0, int lane) {
  const unsigned short* p = W + (size_t)(n0 + (lane & 15)) * ldk + k0 + ((lane & 16) ? 8 : 0);
  v16bf b;
  ((u32x4*)&b)[0] = *(const u32x4*)(p);
  ((u32x4*)&b)[1] = *(const u32x4*)(p + 16);
  return b;
}

__device__ __forceinline__ v8f wmma_bf(v16bf a, v16bf b, v8f c) {
  return __builtin_amdgcn_wmma_f32_16x16x32_bf16(false, a, false, b, (short)0, c, false, false);
}

// ---------------- packing kernels (f32 -> padded bf16) ----------------
__global__ void pack_emb(const float* __restrict__ W, unsigned short* __restrict__ dst) {
  int i = blockIdx.x * 256 + threadIdx.x;
  if (i >= kNTOK * kKP) return;
  int t = i / kKP, k = i % kKP;
  dst[i] = (k < kNINP) ? f2bf(W[(size_t)t * kNINP + k]) : (unsigned short)0;
}
__global__ void pack_wih(const float* __restrict__ W, unsigned short* __restrict__ dst) {
  int i = blockIdx.x * 256 + threadIdx.x;
  if (i >= kHP * kKP) return;
  int h = i / kKP, k = i % kKP;
  dst[i] = (h < kNHID && k < kNINP) ? f2bf(W[(size_t)h * kNINP + k]) : (unsigned short)0;
}
__global__ void pack_whh(const float* __restrict__ W, unsigned short* __restrict__ dst) {
  int i = blockIdx.x * 256 + threadIdx.x;
  if (i >= kHP * kHP) return;
  int h = i / kHP, k = i % kHP;
  dst[i] = (h < kNHID && k < kNHID) ? f2bf(W[(size_t)h * kNHID + k]) : (unsigned short)0;
}
__global__ void pack_h0(const float* __restrict__ hidden, unsigned short* __restrict__ Hbf0,
                        float* __restrict__ ro) {
  int i = blockIdx.x * 256 + threadIdx.x;
  if (i >= kB * kHP) return;
  int b = i / kHP, h = i % kHP;
  float v = (h < kNHID) ? hidden[(size_t)b * kNHID + h] : 0.f;
  ro[i] = v;
  Hbf0[i] = f2bf(v);
}

// ---------------- xW = emb_W[data] @ W_ih^T + b_ih  [8960 x 1152] ----------------
// A tile (16 gathered token rows) shared by all 8 waves -> stage once in LDS.
__global__ void xw_gemm(const int* __restrict__ data, const unsigned short* __restrict__ emb_bf,
                        const unsigned short* __restrict__ Wih, const float* __restrict__ b_ih,
                        float* __restrict__ xW) {
  __shared__ __align__(16) unsigned short sA[16][kKP + 8];
  int tid = threadIdx.x, lane = tid & 31, wave = tid >> 5;
  int mt = blockIdx.x;                 // 0..559
  int ct = blockIdx.y * 8 + wave;      // 0..71
  {
    const int cpr = kKP / 8;           // 52 16-byte chunks per row
    for (int i = tid; i < 16 * cpr; i += 256) {
      int r = i / cpr, c = i % cpr;
      int tok = data[mt * 16 + r];
      *(u32x4*)&sA[r][c * 8] = *(const u32x4*)(emb_bf + (size_t)tok * kKP + c * 8);
    }
  }
  __syncthreads();
  v8f c = {};
  for (int k0 = 0; k0 < kKP; k0 += 32) {
    int ka = k0 + ((lane & 16) ? 8 : 0);
    v16bf a;
    ((u32x4*)&a)[0] = *(const u32x4*)&sA[lane & 15][ka];
    ((u32x4*)&a)[1] = *(const u32x4*)&sA[lane & 15][ka + 16];
    c = wmma_bf(a, load_b(Wih, kKP, ct * 16, k0, lane), c);
  }
  int n = ct * 16 + (lane & 15);
  float bv = (n < kNHID) ? b_ih[n] : 0.f;
  int mh = (lane & 16) ? 8 : 0;
  for (int r = 0; r < 8; ++r)
    xW[(size_t)(mt * 16 + r + mh) * kHP + n] = c[r] + bv;
}

// ---------------- one recurrent step; also stores hU (recurrent pre-activation) ----------------
// B tile (Whh cols, 36 KB) shared by all 8 waves -> stage once in LDS (conflict-free stride).
__global__ void scan_step(int t, const unsigned short* __restrict__ Whh,
                          const unsigned short* __restrict__ HbfIn, unsigned short* __restrict__ HbfOut,
                          const float* __restrict__ xW, const float* __restrict__ b_hh,
                          float* __restrict__ ro, float* __restrict__ hU) {
  __shared__ __align__(16) unsigned short sB[16][kHP + 8];
  int tid = threadIdx.x, lane = tid & 31, wave = tid >> 5; // wave = M-tile (8 tiles of 16 rows)
  int ct = blockIdx.x;                                     // 0..71
  {
    const int cpr = kHP / 8;          // 144 16-byte chunks per row
    for (int i = tid; i < 16 * cpr; i += 256) {
      int nn = i / cpr, c = i % cpr;
      *(u32x4*)&sB[nn][c * 8] = *(const u32x4*)(Whh + (size_t)(ct * 16 + nn) * kHP + c * 8);
    }
  }
  __syncthreads();
  const unsigned short* rowp = HbfIn + (size_t)(wave * 16 + (lane & 15)) * kHP;
  v8f c = {};
  for (int k0 = 0; k0 < kHP; k0 += 32) {
    v16bf a = load_a(rowp, k0, lane);
    int kb = k0 + ((lane & 16) ? 8 : 0);
    v16bf b;
    ((u32x4*)&b)[0] = *(const u32x4*)&sB[lane & 15][kb];
    ((u32x4*)&b)[1] = *(const u32x4*)&sB[lane & 15][kb + 16];
    c = wmma_bf(a, b, c);
  }
  int n = ct * 16 + (lane & 15);
  float bv = (n < kNHID) ? b_hh[n] : 0.f;
  int mh = (lane & 16) ? 8 : 0;
  for (int r = 0; r < 8; ++r) {
    int row = wave * 16 + r + mh;          // 0..127
    size_t pidx = (size_t)(t * kB + row) * kHP + n;
    float pre = c[r] + bv;
    hU[pidx] = pre;                        // reused by negative-sample kernel (saves a 24 GFLOP GEMM)
    float h = tanhf(xW[pidx] + pre);
    ro[(size_t)(t + 1) * kB * kHP + (size_t)row * kHP + n] = h;
    HbfOut[(size_t)row * kHP + n] = f2bf(h);
  }
}

// ---------------- positive distances ----------------
__global__ void pos_kernel(const float* __restrict__ ro, const int* __restrict__ data,
                           const float* __restrict__ bias, float* __restrict__ posarr) {
  int lane = threadIdx.x & 31, wave = threadIdx.x >> 5;
  int pos = blockIdx.x * 8 + wave;                    // 0..8959
  const float* a = ro + (size_t)pos * kHP;            // ro[s][b]
  const float* b = ro + (size_t)(pos + kB) * kHP;     // ro[s+1][b]
  float s = 0.f;
  for (int h = lane; h < kHP; h += 32) { float d = a[h] - b[h]; s += d * d; }
  for (int off = 1; off < 32; off <<= 1) s += __shfl_xor(s, off, 32);
  if (lane == 0) posarr[pos] = TEMPC * (s - bias[data[pos]]);
}

// ---------------- negative samples: fused GEMM + tanh + squared distance ----------------
// A tile staged in LDS once, then hoisted into 13 register fragments per wave and
// reused across all 9 column tiles (eliminates 9x redundant A loads).
__global__ void neg_dist(const int* __restrict__ samples, const unsigned short* __restrict__ emb_bf,
                         const unsigned short* __restrict__ Wih, const float* __restrict__ hU,
                         const float* __restrict__ ro, const float* __restrict__ bias,
                         float* __restrict__ dist) {
  __shared__ __align__(16) unsigned short sA[16][kKP + 8];
  __shared__ float swr[8][16];
  int tid = threadIdx.x, lane = tid & 31, wave = tid >> 5;
  int ns = blockIdx.y;
  int pos0 = blockIdx.x * 16;
  {
    const int cpr = kKP / 8;           // 52 16-byte chunks per row
    for (int i = tid; i < 16 * cpr; i += 256) {
      int r = i / cpr, c = i % cpr;
      int tok = samples[ns * kNPOS + pos0 + r];
      *(u32x4*)&sA[r][c * 8] = *(const u32x4*)(emb_bf + (size_t)tok * kKP + c * 8);
    }
  }
  __syncthreads();
  v16bf aF[kKC];                        // 13 fragments = 104 VGPRs, reused 9x
#pragma unroll
  for (int kk = 0; kk < kKC; ++kk) {
    int ka = kk * 32 + ((lane & 16) ? 8 : 0);
    ((u32x4*)&aF[kk])[0] = *(const u32x4*)&sA[lane & 15][ka];
    ((u32x4*)&aF[kk])[1] = *(const u32x4*)&sA[lane & 15][ka + 16];
  }
  int mh = (lane & 16) ? 8 : 0;
  float racc[8] = {0.f, 0.f, 0.f, 0.f, 0.f, 0.f, 0.f, 0.f};
  for (int ct = wave; ct < kCT; ct += 8) {            // each wave: 9 column tiles
    if (ct + 8 < kCT)                                 // prefetch next B tile row for this lane
      __builtin_prefetch(Wih + (size_t)((ct + 8) * 16 + (lane & 15)) * kKP, 0, 3);
    v8f c = {};
#pragma unroll
    for (int kk = 0; kk < kKC; ++kk)
      c = wmma_bf(aF[kk], load_b(Wih, kKP, ct * 16, kk * 32, lane), c);
    int n = ct * 16 + (lane & 15);
    for (int r = 0; r < 8; ++r) {
      size_t pidx = (size_t)(pos0 + r + mh) * kHP + n;
      float o = tanhf(c[r] + hU[pidx]);               // out = tanh(sW + hU)
      float d = ro[pidx] - o;                         // flat - out (pads are 0-0)
      racc[r] += d * d;
    }
  }
  for (int r = 0; r < 8; ++r)                          // reduce over n within 16-lane halves
    for (int off = 1; off < 16; off <<= 1) racc[r] += __shfl_xor(racc[r], off, 32);
  if ((lane & 15) == 0)
    for (int r = 0; r < 8; ++r) swr[wave][r + mh] = racc[r];
  __syncthreads();
  if (threadIdx.x < 16) {                              // deterministic fixed-order sum
    float s = 0.f;
    for (int w = 0; w < 8; ++w) s += swr[w][threadIdx.x];
    int pos = pos0 + threadIdx.x;
    dist[ns * kNPOS + pos] = s - bias[samples[ns * kNPOS + pos]];
  }
}

// ---------------- final scalar loss (single block => deterministic) ----------------
__global__ void final_reduce(const float* __restrict__ posarr, const float* __restrict__ dist,
                             const float* __restrict__ bias, float* __restrict__ out) {
  __shared__ float red[1024];
  float s_pl = 0.f, s_b = 0.f;
  for (int i = threadIdx.x; i < kNPOS; i += 1024) {
    float p = posarr[i];
    float se = __expf(-p);
    for (int ns = 0; ns < kNS; ++ns) se += __expf(-TEMPC * dist[ns * kNPOS + i]);
    s_pl += p + __logf(se + EPSL);
  }
  for (int i = threadIdx.x; i < kNTOK; i += 1024) { float b = bias[i]; s_b += b * b; }
  red[threadIdx.x] = s_pl * (1.f / (float)kNPOS) + s_b;
  __syncthreads();
  for (int st = 512; st > 0; st >>= 1) {
    if ((int)threadIdx.x < st) red[threadIdx.x] += red[threadIdx.x + st];
    __syncthreads();
  }
  if (threadIdx.x == 0) out[0] = red[0];
}

__global__ void copy_hidden(const float* __restrict__ ro, float* __restrict__ out) {
  int i = blockIdx.x * 256 + threadIdx.x;
  if (i >= kB * kNHID) return;
  int b = i / kNHID, h = i % kNHID;
  out[1 + i] = ro[(size_t)kS * kB * kHP + (size_t)b * kHP + h];
}

extern "C" void kernel_launch(void* const* d_in, const int* in_sizes, int n_in,
                              void* d_out, int out_size, void* d_ws, size_t ws_size,
                              hipStream_t stream) {
  (void)in_sizes; (void)n_in; (void)out_size; (void)ws_size;
  const int*   data    = (const int*)  d_in[0];
  const float* hidden  = (const float*)d_in[1];
  const int*   samples = (const int*)  d_in[2];
  const float* emb_W   = (const float*)d_in[3];
  const float* W_ih    = (const float*)d_in[4];
  const float* b_ih    = (const float*)d_in[5];
  const float* W_hh    = (const float*)d_in[6];
  const float* b_hh    = (const float*)d_in[7];
  const float* bias    = (const float*)d_in[8];
  float* out = (float*)d_out;

  char* w = (char*)d_ws;
  auto take = [&](size_t bytes) { char* p = w; w += (bytes + 255) & ~(size_t)255; return p; };
  unsigned short* emb_bf = (unsigned short*)take((size_t)kNTOK * kKP * 2); // 27.7 MB
  unsigned short* Wih_bf = (unsigned short*)take((size_t)kHP * kKP * 2);   // 0.96 MB
  unsigned short* Whh_bf = (unsigned short*)take((size_t)kHP * kHP * 2);   // 2.65 MB
  unsigned short* HbfA   = (unsigned short*)take((size_t)kB * kHP * 2);
  unsigned short* HbfB   = (unsigned short*)take((size_t)kB * kHP * 2);
  float* xW   = (float*)take((size_t)kNPOS * kHP * 4);                     // 41.3 MB
  float* ro   = (float*)take((size_t)(kS + 1) * kB * kHP * 4);             // 41.9 MB
  float* hU   = (float*)take((size_t)kNPOS * kHP * 4);                     // 41.3 MB
  float* dist = (float*)take((size_t)kNS * kNPOS * 4);
  float* parr = (float*)take((size_t)kNPOS * 4);

  pack_emb<<<(kNTOK * kKP + 255) / 256, 256, 0, stream>>>(emb_W, emb_bf);
  pack_wih<<<(kHP * kKP + 255) / 256, 256, 0, stream>>>(W_ih, Wih_bf);
  pack_whh<<<(kHP * kHP + 255) / 256, 256, 0, stream>>>(W_hh, Whh_bf);
  pack_h0<<<(kB * kHP + 255) / 256, 256, 0, stream>>>(hidden, HbfA, ro);

  xw_gemm<<<dim3(kMT, 9), 256, 0, stream>>>(data, emb_bf, Wih_bf, b_ih, xW);

  for (int t = 0; t < kS; ++t) {
    const unsigned short* hin = (t & 1) ? HbfB : HbfA;
    unsigned short*      hout = (t & 1) ? HbfA : HbfB;
    scan_step<<<kCT, 256, 0, stream>>>(t, Whh_bf, hin, hout, xW, b_hh, ro, hU);
  }

  pos_kernel<<<kNPOS / 8, 256, 0, stream>>>(ro, data, bias, parr);
  neg_dist<<<dim3(kMT, kNS), 256, 0, stream>>>(samples, emb_bf, Wih_bf, hU, ro, bias, dist);
  final_reduce<<<1, 1024, 0, stream>>>(parr, dist, bias, out);
  copy_hidden<<<(kB * kNHID + 255) / 256, 256, 0, stream>>>(ro, out);
}
